// GraphormerLayer_17205638988080
// MI455X (gfx1250) — compile-verified, matrix operand order checked by reading, not run on hardware
//
#include <hip/hip_runtime.h>
#include <hip/hip_bf16.h>

typedef __attribute__((ext_vector_type(16))) _Float16 v16h;
typedef __attribute__((ext_vector_type(8)))  float    v8f;

#define LDA   40            // padded LDS row stride (halves); 80B rows -> 8B-aligned float4 slots
#define PLANE (64 * LDA)    // offset of the "lo" f16 plane
#define BUFSZ (2 * PLANE)   // one double-buffer slot (hi+lo planes)

// ---------------------------------------------------------------------------
// Tiled GEMM with split-f16 WMMA (fp32-accurate):
//   C[M,Nn] = act(A[M,K] @ B[K,Nn] + bias)
// a = hi + lo (both f16);  a*b ~= hi*hi + hi*lo + lo*hi  (3 WMMA issues/tile).
// Block = 256 threads (8 waves). Block tile 64x64, BK = 32, LDS double-buffered.
// Wave (wM 0..3, wN 0..1) computes rows [wM*16,+16) x cols [wN*32,+32).
// ---------------------------------------------------------------------------
__global__ __launch_bounds__(256) void gemm_wmma_f16x2(
    const float* __restrict__ A, const float* __restrict__ B,
    const float* __restrict__ bias, float* __restrict__ C,
    int M, int K, int Nn, int doRelu)
{
    __shared__ _Float16 sA[2 * BUFSZ];   // [buf][hi|lo][m][k]
    __shared__ _Float16 sB[2 * BUFSZ];   // [buf][hi|lo][n][k]  (transposed B)

    const int t    = threadIdx.x;
    const int lane = t & 31;
    const int w    = t >> 5;
    const int wM   = w & 3;
    const int wN   = w >> 2;
    const int rowBase = blockIdx.y * 64;
    const int colBase = blockIdx.x * 64;
    const int r16 = lane & 15;          // M (A,C) / N (B) index within tile
    const int kh  = lane >> 4;          // K-half select per ISA A-layout

    // ---- per-thread staging coordinates (float4 granularity) ----
    // A tile 64x32: 512 float4s; thread handles f = t, t+256
    int mA0 = t >> 3,          cA0 = (t & 7) * 4;
    int mA1 = (t + 256) >> 3,  cA1 = ((t + 256) & 7) * 4;
    // B tile 32(k) x 64(n): thread handles float4 along n
    int kB0 = t >> 4,          nB0 = (t & 15) * 4;
    int kB1 = (t + 256) >> 4,  nB1 = ((t + 256) & 15) * 4;

    const float* aRun0 = A + (size_t)(rowBase + mA0) * K + cA0;
    const float* aRun1 = A + (size_t)(rowBase + mA1) * K + cA1;
    const float* bRun0 = B + (size_t)kB0 * Nn + colBase + nB0;
    const float* bRun1 = B + (size_t)kB1 * Nn + colBase + nB1;
    const size_t bStep = (size_t)32 * Nn;

    union Pack4 { _Float16 h[4]; uint2 u; };

    // Stage one K-tile into buffer `buf`, advance running pointers.
    auto stage = [&](int buf) {
        _Float16* dA = &sA[buf * BUFSZ];
        _Float16* dB = &sB[buf * BUFSZ];
        // --- A: 2 x float4, packed ds_store_b64 per plane ---
        {
            float4 va = *(const float4*)aRun0;
            float4 vb = *(const float4*)aRun1;
            __builtin_prefetch(aRun0 + 32, 0, 3);   // WGP-scope prefetch
            __builtin_prefetch(aRun1 + 32, 0, 3);
            aRun0 += 32; aRun1 += 32;
            Pack4 h, l;
            h.h[0] = (_Float16)va.x; l.h[0] = (_Float16)(va.x - (float)h.h[0]);
            h.h[1] = (_Float16)va.y; l.h[1] = (_Float16)(va.y - (float)h.h[1]);
            h.h[2] = (_Float16)va.z; l.h[2] = (_Float16)(va.z - (float)h.h[2]);
            h.h[3] = (_Float16)va.w; l.h[3] = (_Float16)(va.w - (float)h.h[3]);
            *(uint2*)&dA[mA0 * LDA + cA0]         = h.u;
            *(uint2*)&dA[PLANE + mA0 * LDA + cA0] = l.u;
            h.h[0] = (_Float16)vb.x; l.h[0] = (_Float16)(vb.x - (float)h.h[0]);
            h.h[1] = (_Float16)vb.y; l.h[1] = (_Float16)(vb.y - (float)h.h[1]);
            h.h[2] = (_Float16)vb.z; l.h[2] = (_Float16)(vb.z - (float)h.h[2]);
            h.h[3] = (_Float16)vb.w; l.h[3] = (_Float16)(vb.w - (float)h.h[3]);
            *(uint2*)&dA[mA1 * LDA + cA1]         = h.u;
            *(uint2*)&dA[PLANE + mA1 * LDA + cA1] = l.u;
        }
        // --- B: 2 x float4 along n, transposed scatter into [n][k] ---
        {
            float4 va = *(const float4*)bRun0;
            float4 vb = *(const float4*)bRun1;
            __builtin_prefetch(bRun0 + bStep, 0, 3);
            __builtin_prefetch(bRun1 + bStep, 0, 3);
            bRun0 += bStep; bRun1 += bStep;
            float e[4] = {va.x, va.y, va.z, va.w};
            #pragma unroll
            for (int jn = 0; jn < 4; ++jn) {
                _Float16 hi = (_Float16)e[jn];
                _Float16 lo = (_Float16)(e[jn] - (float)hi);
                dB[(nB0 + jn) * LDA + kB0]         = hi;
                dB[PLANE + (nB0 + jn) * LDA + kB0] = lo;
            }
            float f[4] = {vb.x, vb.y, vb.z, vb.w};
            #pragma unroll
            for (int jn = 0; jn < 4; ++jn) {
                _Float16 hi = (_Float16)f[jn];
                _Float16 lo = (_Float16)(f[jn] - (float)hi);
                dB[(nB1 + jn) * LDA + kB1]         = hi;
                dB[PLANE + (nB1 + jn) * LDA + kB1] = lo;
            }
        }
    };

    v8f acc0 = {};
    v8f acc1 = {};

    stage(0);                           // prologue: tile 0

    int buf = 0;
    for (int k0 = 0; k0 < K; k0 += 32) {
        __syncthreads();                // staging of `buf` complete
        if (k0 + 32 < K) stage(buf ^ 1);// overlap next tile with WMMA chain

        const _Float16* bA = &sA[buf * BUFSZ];
        const _Float16* bB = &sB[buf * BUFSZ];
        // Fragments per CDNA5 ISA 16-bit 16x32 layout:
        //  lane L<16: VGPR p<4 holds K=2p,2p+1 ; p>=4 holds K=16+2(p-4),..
        //  lane L>=16: same + 8
        union { v16h h; unsigned u[8]; } afh, afl, b0h, b0l, b1h, b1l;
        #pragma unroll
        for (int p = 0; p < 8; ++p) {
            int kb = ((p & 4) ? 16 : 0) + kh * 8 + (p & 3) * 2;
            int ia  = (wM * 16 + r16) * LDA + kb;
            int ib0 = (wN * 32 + r16) * LDA + kb;
            int ib1 = (wN * 32 + 16 + r16) * LDA + kb;
            afh.u[p] = *(const unsigned*)&bA[ia];
            afl.u[p] = *(const unsigned*)&bA[PLANE + ia];
            b0h.u[p] = *(const unsigned*)&bB[ib0];
            b0l.u[p] = *(const unsigned*)&bB[PLANE + ib0];
            b1h.u[p] = *(const unsigned*)&bB[ib1];
            b1l.u[p] = *(const unsigned*)&bB[PLANE + ib1];
        }
        acc0 = __builtin_amdgcn_wmma_f32_16x16x32_f16(false, afh.h, false, b0h.h, (short)0, acc0, false, false);
        acc0 = __builtin_amdgcn_wmma_f32_16x16x32_f16(false, afh.h, false, b0l.h, (short)0, acc0, false, false);
        acc0 = __builtin_amdgcn_wmma_f32_16x16x32_f16(false, afl.h, false, b0h.h, (short)0, acc0, false, false);
        acc1 = __builtin_amdgcn_wmma_f32_16x16x32_f16(false, afh.h, false, b1h.h, (short)0, acc1, false, false);
        acc1 = __builtin_amdgcn_wmma_f32_16x16x32_f16(false, afh.h, false, b1l.h, (short)0, acc1, false, false);
        acc1 = __builtin_amdgcn_wmma_f32_16x16x32_f16(false, afl.h, false, b1h.h, (short)0, acc1, false, false);
        buf ^= 1;
    }

    // Epilogue: C/D layout — elem r of lane L -> row r + 8*(L/16), col L%16
    #pragma unroll
    for (int r = 0; r < 8; ++r) {
        int row  = rowBase + wM * 16 + kh * 8 + r;
        int col0 = colBase + wN * 32 + r16;
        float v0 = acc0[r] + bias[col0];
        float v1 = acc1[r] + bias[col0 + 16];
        if (doRelu) { v0 = fmaxf(v0, 0.f); v1 = fmaxf(v1, 0.f); }
        C[(size_t)row * Nn + col0]      = v0;
        C[(size_t)row * Nn + col0 + 16] = v1;
    }
}

// ---------------------------------------------------------------------------
// Sparse graph attention. One block per node, 4 waves = 4 heads.
// Lane j owns edge e = n*32 + j (edges are grouped per source node).
// ---------------------------------------------------------------------------
__global__ __launch_bounds__(128) void attn_kernel(
    const float* __restrict__ q, const float* __restrict__ k,
    const float* __restrict__ v, const int* __restrict__ ei,
    const float* __restrict__ ea, const float* __restrict__ We,
    const float* __restrict__ be, float* __restrict__ out, int Nnodes)
{
    const int n = blockIdx.x;
    const int h = threadIdx.x >> 5;
    const int j = threadIdx.x & 31;
    const int E = Nnodes * 32;
    const int e = n * 32 + j;
    const int srcN = ei[e];
    const int dstN = ei[E + e];

    const float* qp = q + (size_t)srcN * 256 + h * 64;
    const float* kp = k + (size_t)dstN * 256 + h * 64;
    float dot = 0.f;
    #pragma unroll
    for (int d = 0; d < 64; d += 4) {
        float4 qv = *(const float4*)(qp + d);
        float4 kv = *(const float4*)(kp + d);
        dot += qv.x * kv.x + qv.y * kv.y + qv.z * kv.z + qv.w * kv.w;
    }
    float eb = be[h];
    #pragma unroll
    for (int c = 0; c < 16; ++c)
        eb += ea[(size_t)e * 16 + c] * We[c * 4 + h];
    float logit = dot * 0.125f + eb;   // 1/sqrt(64)

    float m = logit;
    #pragma unroll
    for (int s = 16; s > 0; s >>= 1) m = fmaxf(m, __shfl_xor(m, s, 32));
    float p = __expf(logit - m);
    float sum = p;
    #pragma unroll
    for (int s = 16; s > 0; s >>= 1) sum += __shfl_xor(sum, s, 32);
    float a = p / sum;

    __shared__ float s_attn[4][32];
    __shared__ int   s_dst[4][32];
    s_attn[h][j] = a;
    s_dst[h][j]  = dstN;
    __syncthreads();

    const int d0 = j * 2;
    float acc0 = 0.f, acc1 = 0.f;
    #pragma unroll 4
    for (int jj = 0; jj < 32; ++jj) {
        float aw = s_attn[h][jj];
        const float* vp = v + (size_t)s_dst[h][jj] * 256 + h * 64 + d0;
        acc0 += aw * vp[0];
        acc1 += aw * vp[1];
    }
    out[(size_t)n * 256 + h * 64 + d0]     = acc0;
    out[(size_t)n * 256 + h * 64 + d0 + 1] = acc1;
}

// ---------------------------------------------------------------------------
// Fused residual + LayerNorm over HID=256. One block (256 thr) per node.
// ---------------------------------------------------------------------------
__global__ __launch_bounds__(256) void ln_kernel(
    const float* __restrict__ a, const float* __restrict__ b,
    const float* __restrict__ g, const float* __restrict__ beta,
    float* __restrict__ out)
{
    const int n = blockIdx.x;
    const int t = threadIdx.x;
    float val = a[(size_t)n * 256 + t] + b[(size_t)n * 256 + t];

    __shared__ float red[2][8];
    float s = val, s2 = val * val;
    #pragma unroll
    for (int off = 16; off > 0; off >>= 1) {
        s  += __shfl_xor(s,  off, 32);
        s2 += __shfl_xor(s2, off, 32);
    }
    const int w = t >> 5, lane = t & 31;
    if (lane == 0) { red[0][w] = s; red[1][w] = s2; }
    __syncthreads();
    if (w == 0) {
        float ts  = (lane < 8) ? red[0][lane] : 0.f;
        float ts2 = (lane < 8) ? red[1][lane] : 0.f;
        #pragma unroll
        for (int off = 4; off > 0; off >>= 1) {
            ts  += __shfl_xor(ts,  off, 32);
            ts2 += __shfl_xor(ts2, off, 32);
        }
        if (lane == 0) { red[0][0] = ts; red[1][0] = ts2; }
    }
    __syncthreads();
    float mean = red[0][0] * (1.f / 256.f);
    float var  = red[1][0] * (1.f / 256.f) - mean * mean;
    out[(size_t)n * 256 + t] =
        (val - mean) * rsqrtf(var + 1e-5f) * g[t] + beta[t];
}

// ---------------------------------------------------------------------------
extern "C" void kernel_launch(void* const* d_in, const int* in_sizes, int n_in,
                              void* d_out, int out_size, void* d_ws, size_t ws_size,
                              hipStream_t stream) {
    const float* x   = (const float*)d_in[0];
    const int*   ei  = (const int*)  d_in[1];
    const float* ea  = (const float*)d_in[2];
    const float* Wq  = (const float*)d_in[3];
    const float* bq  = (const float*)d_in[4];
    const float* Wk  = (const float*)d_in[5];
    const float* bk  = (const float*)d_in[6];
    const float* Wv  = (const float*)d_in[7];
    const float* bv  = (const float*)d_in[8];
    const float* Wo  = (const float*)d_in[9];
    const float* bo  = (const float*)d_in[10];
    const float* We  = (const float*)d_in[11];
    const float* be  = (const float*)d_in[12];
    const float* g1  = (const float*)d_in[13];
    const float* b1  = (const float*)d_in[14];
    const float* g2  = (const float*)d_in[15];
    const float* b2  = (const float*)d_in[16];
    const float* Wf1 = (const float*)d_in[17];
    const float* bf1 = (const float*)d_in[18];
    const float* Wf2 = (const float*)d_in[19];
    const float* bf2 = (const float*)d_in[20];

    const int N = 4096, HID = 256;
    const size_t S = (size_t)N * HID;
    float* ws = (float*)d_ws;
    float* q    = ws + 0 * S;
    float* kbuf = ws + 1 * S;
    float* vbuf = ws + 2 * S;
    float* attn = ws + 3 * S;
    float* ffn  = ws + 4 * S;   // N x 1024 -> occupies slots 4..7
    float* o    = q;            // reuse (q dead after attention)
    float* x1   = vbuf;         // reuse (v dead after attention)
    float* y    = kbuf;         // reuse (k dead after attention)

    dim3 blk(256);
    // Q / K / V projections
    gemm_wmma_f16x2<<<dim3(4, 64), blk, 0, stream>>>(x, Wq, bq, q,    N, 256, 256, 0);
    gemm_wmma_f16x2<<<dim3(4, 64), blk, 0, stream>>>(x, Wk, bk, kbuf, N, 256, 256, 0);
    gemm_wmma_f16x2<<<dim3(4, 64), blk, 0, stream>>>(x, Wv, bv, vbuf, N, 256, 256, 0);
    // Sparse graph attention (softmax over DEG=32 neighbors per head)
    attn_kernel<<<N, 128, 0, stream>>>(q, kbuf, vbuf, ei, ea, We, be, attn, N);
    // Output projection, residual + LN1
    gemm_wmma_f16x2<<<dim3(4, 64), blk, 0, stream>>>(attn, Wo, bo, o, N, 256, 256, 0);
    ln_kernel<<<N, 256, 0, stream>>>(x, o, g1, b1, x1);
    // FFN
    gemm_wmma_f16x2<<<dim3(16, 64), blk, 0, stream>>>(x1, Wf1, bf1, ffn, N, 256, 1024, 1);
    gemm_wmma_f16x2<<<dim3(4, 64),  blk, 0, stream>>>(ffn, Wf2, bf2, y, N, 1024, 256, 0);
    // Residual + LN2 -> output
    ln_kernel<<<N, 256, 0, stream>>>(x1, y, g2, b2, (float*)d_out);
}
